// NeuralAdditiveModel_23691039604774
// MI455X (gfx1250) — matrix operand order repaired and checked by reading, not compiled, not run on hardware
//
#include <hip/hip_runtime.h>

#define F_DIM 256
#define H1 128
#define H2 64
#define H3 32
#define BATCH 8192

typedef __attribute__((ext_vector_type(16))) __bf16 v16bf;
typedef __attribute__((ext_vector_type(8)))  float  v8f;

__device__ __forceinline__ float relu_f(float v) { return v > 0.0f ? v : 0.0f; }

// A-fragment K-slot mapping (16-bit A 16x32, ISA 7.12.2): VGPR j holds a K-pair.
// lanes 0-15: K = {0..7, 16..23}; lanes 16-31: K = {8..15, 24..31}
__device__ __forceinline__ int kslotA(int j, int h) {
  return (j < 4) ? (2 * j + 8 * h) : (16 + 2 * (j - 4) + 8 * h);
}

__global__ void nam_init_out(float* __restrict__ out, const float* __restrict__ bias) {
  int i = blockIdx.x * blockDim.x + threadIdx.x;
  if (i < BATCH) out[i] = bias[0];
}

__global__ __launch_bounds__(128) void nam_feature_mlp(
    const float* __restrict__ x,  const float* __restrict__ W1, const float* __restrict__ b1,
    const float* __restrict__ W2, const float* __restrict__ b2,
    const float* __restrict__ W3, const float* __restrict__ b3,
    const float* __restrict__ W4, const float* __restrict__ b4,
    float* __restrict__ out)
{
  __shared__ __align__(16) float sW1[H1];
  __shared__ __align__(16) float sB1[H1];
  // B-fragment-swizzled weights: [kTile][nTile][lane] = 16 bf16 = one 32B vector per lane
  __shared__ v16bf sW2f[4][4][32];   // 16 KB
  __shared__ float sB2[H2];
  __shared__ v16bf sW3f[2][2][32];   // 4 KB
  __shared__ float sB3[H3];
  __shared__ __align__(16) float sW4[H3];
  __shared__ float sB4[1];
  // per-wave h2 staging, A-fragment-swizzled: [wave][kTile][lane]
  __shared__ v16bf sH2f[4][2][32];   // 8 KB
  __shared__ __align__(16) float sH3[4][16 * H3]; // 8 KB

  const int tid  = threadIdx.x;
  const int f    = blockIdx.x;             // feature id
  const int rowBlock = blockIdx.y * 256;   // 256 batch rows per block
  const int w    = tid >> 5;               // wave id (0..3)
  const int lane = tid & 31;
  const int half = lane >> 4;
  const int rc   = lane & 15;

  // ---- Stage this feature's weights into LDS (f32->bf16, pre-swizzled to frag layout) ----
  if (tid < H1) { sW1[tid] = W1[f * H1 + tid]; sB1[tid] = b1[f * H1 + tid]; }
  {
    const float4* p = (const float4*)(W2 + (size_t)f * (H1 * H2));
    for (int i = tid; i < (H1 * H2) / 4; i += 128) {
      float4 v = p[i];
      int flat = 4 * i;
      int k  = flat >> 6;          // row (K index), 64 floats per row
      int n0 = flat & 63;          // starting column
      int kt = k >> 5, kl = k & 31;
      int lh   = ((kl >> 3) & 1) << 4;   // B-frag: half = (kl/8)%2
      int slot = (kl & 7) | ((kl >> 4) << 3);  // B-frag: vector slot = kl%8 + 8*(kl/16)
      int nt = n0 >> 4, nr = n0 & 15;
      __bf16* dst = (__bf16*)&sW2f[kt][nt][0];
      dst[(lh + nr + 0) * 16 + slot] = (__bf16)v.x;
      dst[(lh + nr + 1) * 16 + slot] = (__bf16)v.y;
      dst[(lh + nr + 2) * 16 + slot] = (__bf16)v.z;
      dst[(lh + nr + 3) * 16 + slot] = (__bf16)v.w;
    }
  }
  {
    const float4* p = (const float4*)(W3 + (size_t)f * (H2 * H3));
    for (int i = tid; i < (H2 * H3) / 4; i += 128) {
      float4 v = p[i];
      int flat = 4 * i;
      int k  = flat >> 5;          // 32 floats per row
      int n0 = flat & 31;
      int kt = k >> 5, kl = k & 31;
      int lh   = ((kl >> 3) & 1) << 4;
      int slot = (kl & 7) | ((kl >> 4) << 3);
      int nt = n0 >> 4, nr = n0 & 15;
      __bf16* dst = (__bf16*)&sW3f[kt][nt][0];
      dst[(lh + nr + 0) * 16 + slot] = (__bf16)v.x;
      dst[(lh + nr + 1) * 16 + slot] = (__bf16)v.y;
      dst[(lh + nr + 2) * 16 + slot] = (__bf16)v.z;
      dst[(lh + nr + 3) * 16 + slot] = (__bf16)v.w;
    }
  }
  if (tid < H2) sB2[tid] = b2[f * H2 + tid];
  if (tid < H3) { sB3[tid] = b3[f * H3 + tid]; sW4[tid] = W4[f * H3 + tid]; }
  if (tid == 0) sB4[0] = b4[f];
  __syncthreads();

  v8f zero8;
#pragma unroll
  for (int i = 0; i < 8; ++i) zero8[i] = 0.0f;

  // Layer-3 B fragments are invariant across batch tiles: hoist into registers (32 VGPRs)
  v16bf w3b[2][2];
#pragma unroll
  for (int kt = 0; kt < 2; ++kt)
#pragma unroll
    for (int nt = 0; nt < 2; ++nt) w3b[kt][nt] = sW3f[kt][nt][lane];

  // ---- 4 tiles of 16 batch rows per wave ----
  for (int it = 0; it < 4; ++it) {
    const int rowBase = rowBlock + w * 64 + it * 16;
    const float xm = x[(size_t)(rowBase + rc) * F_DIM + f];

    // ======== Layer 1 (registers) + Layer 2 (WMMA, 16x128 @ 128x64) ========
    v8f acc2[4];
#pragma unroll
    for (int nt = 0; nt < 4; ++nt) acc2[nt] = zero8;

#pragma unroll
    for (int kt = 0; kt < 4; ++kt) {
      const int k0 = kt * 32;
      v16bf a;
#pragma unroll
      for (int j = 0; j < 8; ++j) {
        int k = k0 + kslotA(j, half);
        a[2 * j]     = (__bf16)relu_f(fmaf(xm, sW1[k],     sB1[k]));
        a[2 * j + 1] = (__bf16)relu_f(fmaf(xm, sW1[k + 1], sB1[k + 1]));
      }
#pragma unroll
      for (int nt = 0; nt < 4; ++nt) {
        v16bf bm = sW2f[kt][nt][lane];   // single 32B vector load per fragment
        acc2[nt] = __builtin_amdgcn_wmma_f32_16x16x32_bf16(
            false, a, false, bm, (short)0, acc2[nt], false, false);
      }
    }

    // bias + relu; scatter h2 straight into A-fragment-swizzled LDS for layer 3
#pragma unroll
    for (int nt = 0; nt < 4; ++nt) {
      const int kk = nt * 16 + rc;           // layer-3 K index
      const int kt2 = kk >> 5, kl = kk & 31;
      const int hh   = (kl >> 3) & 1;        // A-frag half
      const int slot = (kl & 7) | ((kl >> 4) << 3);
      __bf16* dst = (__bf16*)&sH2f[w][kt2][0];
#pragma unroll
      for (int r = 0; r < 8; ++r) {
        int m = r + 8 * half;                // C-frag row
        dst[(hh * 16 + m) * 16 + slot] = (__bf16)relu_f(acc2[nt][r] + sB2[kk]);
      }
    }
    asm volatile("s_wait_dscnt 0" ::: "memory");

    // ======== Layer 3 (WMMA, 16x64 @ 64x32) ========
    v8f acc3[2];
#pragma unroll
    for (int nt = 0; nt < 2; ++nt) acc3[nt] = zero8;

#pragma unroll
    for (int kt = 0; kt < 2; ++kt) {
      v16bf a2 = sH2f[w][kt][lane];          // single 32B vector load
#pragma unroll
      for (int nt = 0; nt < 2; ++nt) {
        acc3[nt] = __builtin_amdgcn_wmma_f32_16x16x32_bf16(
            false, a2, false, w3b[kt][nt], (short)0, acc3[nt], false, false);
      }
    }

    // bias + relu -> sH3 (f32, row-major 16x32 per wave)
#pragma unroll
    for (int nt = 0; nt < 2; ++nt) {
#pragma unroll
      for (int r = 0; r < 8; ++r) {
        int m = r + 8 * half;
        int n = nt * 16 + rc;
        sH3[w][m * H3 + n] = relu_f(acc3[nt][r] + sB3[n]);
      }
    }
    asm volatile("s_wait_dscnt 0" ::: "memory");

    // ======== Layer 4 (H3 -> 1): vectorized dot + half-wave reduce + atomic ========
    float partial = 0.0f;
    const float4* h3v = (const float4*)&sH3[w][rc * H3 + half * 16];
    const float4* w4v = (const float4*)&sW4[half * 16];
#pragma unroll
    for (int q = 0; q < 4; ++q) {
      float4 hv = h3v[q];
      float4 wv = w4v[q];
      partial = fmaf(hv.x, wv.x, partial);
      partial = fmaf(hv.y, wv.y, partial);
      partial = fmaf(hv.z, wv.z, partial);
      partial = fmaf(hv.w, wv.w, partial);
    }
    float other = __shfl_xor(partial, 16, 32);
    if (lane < 16) {
      float v = partial + other + sB4[0];
      unsafeAtomicAdd(&out[rowBase + rc], v);  // global_atomic_add_f32
    }
  }
}

extern "C" void kernel_launch(void* const* d_in, const int* in_sizes, int n_in,
                              void* d_out, int out_size, void* d_ws, size_t ws_size,
                              hipStream_t stream) {
  const float* x    = (const float*)d_in[0];
  const float* W1   = (const float*)d_in[1];
  const float* b1   = (const float*)d_in[2];
  const float* W2   = (const float*)d_in[3];
  const float* b2   = (const float*)d_in[4];
  const float* W3   = (const float*)d_in[5];
  const float* b3   = (const float*)d_in[6];
  const float* W4   = (const float*)d_in[7];
  const float* b4   = (const float*)d_in[8];
  const float* bias = (const float*)d_in[9];
  float* out = (float*)d_out;

  nam_init_out<<<BATCH / 256, 256, 0, stream>>>(out, bias);
  dim3 grid(F_DIM, BATCH / 256);
  nam_feature_mlp<<<grid, 128, 0, stream>>>(x, W1, b1, W2, b2, W3, b3, W4, b4, out);
}
